// MambaLayer_56332791054515
// MI455X (gfx1250) — matrix operand order repaired and checked by reading, not compiled, not run on hardware
//
#include <hip/hip_runtime.h>
#include <math.h>

// ---------------- problem constants (from reference) ----------------
#define NB   2            // batch
#define NC   384          // DIM
#define NL   4096         // tokens (16*16*16)
#define NDI  768          // D_INNER
#define NDS  16           // D_STATE
#define NDTR 24           // DT_RANK
#define NSL  16           // num slices
#define LSL  256          // NL/NSL
#define NTOK (NB*NL)      // 8192 rows for token GEMMs

typedef unsigned short u16t;
typedef __attribute__((ext_vector_type(16))) __bf16 v16bf;
typedef __attribute__((ext_vector_type(8)))  float  v8f;

union FragAB { v16bf v; uint4 q[2]; };

__device__ __forceinline__ u16t f2bf(float f) {
  unsigned u = __float_as_uint(f);
  u += 0x7FFFu + ((u >> 16) & 1u);          // round-to-nearest-even
  return (u16t)(u >> 16);
}
__device__ __forceinline__ float bf2f(u16t h) {
  return __uint_as_float(((unsigned)h) << 16);
}
__device__ __forceinline__ float siluf(float v) { return v / (1.0f + __expf(-v)); }
__device__ __forceinline__ float softplusf(float v) {
  return (v > 20.0f) ? v : log1pf(__expf(v));
}
__device__ __forceinline__ float sigmoidf(float v) { return 1.0f / (1.0f + __expf(-v)); }

// gfx1250 async memory->LDS copy (ASYNCcnt-tracked, CDNA5 ISA)
__device__ __forceinline__ void async_ldst_b128(unsigned lds_off, const u16t* g) {
  asm volatile("global_load_async_to_lds_b128 %0, %1, off"
               :: "v"(lds_off), "v"((unsigned long long)(size_t)g)
               : "memory");
}

// ---------------- packing kernels ----------------
__global__ __launch_bounds__(256)
void pack_bf16_kernel(const float* __restrict__ src, u16t* __restrict__ dst, int n) {
  int i = blockIdx.x * blockDim.x + threadIdx.x;
  if (i < n) dst[i] = f2bf(src[i]);
}

// src [srcR, srcC] f32 -> dst [dstR, dstC] bf16, zero padded
__global__ __launch_bounds__(256)
void pack_pad_bf16_kernel(const float* __restrict__ src, u16t* __restrict__ dst,
                          int srcR, int srcC, int dstR, int dstC) {
  int i = blockIdx.x * blockDim.x + threadIdx.x;
  if (i >= dstR * dstC) return;
  int r = i / dstC, c = i % dstC;
  float v = (r < srcR && c < srcC) ? src[(size_t)r * srcC + c] : 0.0f;
  dst[i] = f2bf(v);
}

// take dt_r columns [0,24) of dbl rows (stride 64), pad K to 32, bf16
__global__ __launch_bounds__(256)
void pack_dtr_kernel(const float* __restrict__ dbl, u16t* __restrict__ dst, int M) {
  int i = blockIdx.x * blockDim.x + threadIdx.x;
  if (i >= M * 32) return;
  int m = i >> 5, c = i & 31;
  dst[i] = f2bf(c < NDTR ? dbl[(size_t)m * 64 + c] : 0.0f);
}

// ---------------- LayerNorm: x[B,C,L] -> xn_bf[B*L, C] ----------------
__global__ __launch_bounds__(128)
void layernorm_kernel(const float* __restrict__ x, const float* __restrict__ g,
                      const float* __restrict__ beta, u16t* __restrict__ xn) {
  int token = blockIdx.x;                 // b*NL + l
  int b = token >> 12, l = token & (NL - 1);
  const float* xp = x + (size_t)b * NC * NL + l;
  __shared__ float r1[128], r2[128];
  float s = 0.f, s2 = 0.f;
  for (int c = threadIdx.x; c < NC; c += 128) {
    float v = xp[(size_t)c * NL]; s += v; s2 += v * v;
  }
  r1[threadIdx.x] = s; r2[threadIdx.x] = s2; __syncthreads();
  for (int st = 64; st > 0; st >>= 1) {
    if ((int)threadIdx.x < st) { r1[threadIdx.x] += r1[threadIdx.x + st]; r2[threadIdx.x] += r2[threadIdx.x + st]; }
    __syncthreads();
  }
  float mean = r1[0] * (1.0f / NC);
  float var  = r2[0] * (1.0f / NC) - mean * mean;
  float rstd = rsqrtf(var + 1e-5f);
  u16t* op = xn + (size_t)token * NC;
  for (int c = threadIdx.x; c < NC; c += 128)
    op[c] = f2bf((xp[(size_t)c * NL] - mean) * rstd * g[c] + beta[c]);
}

// ---------------- WMMA bf16 GEMM: C[M,N] = A[M,K] * B[N,K]^T ----------------
// Block = 8 waves = 128(M) x 64(N) tile. B tile staged to LDS via
// global_load_async_to_lds_b128 (double buffered, ASYNCcnt), A fragment
// double-buffered in registers so WMMAs overlap the next stage's loads.
// act: 0=none, 1=relu, 2=softplus
__global__ __launch_bounds__(256)
void wmma_gemm_kernel(const u16t* __restrict__ A, const u16t* __restrict__ B,
                      const float* __restrict__ bias, float* __restrict__ C,
                      u16t* __restrict__ Cbf, int M, int N, int K, int act) {
  __shared__ u16t smem[2][64 * 32];       // 2 x 4KB B tiles (64 cols x 32 K)
  const int tid  = threadIdx.x;
  const int lane = tid & 31;
  const int wv   = tid >> 5;              // 0..7
  const int n64  = N >> 6;
  const int bm   = blockIdx.x / n64;
  const int bn   = blockIdx.x % n64;
  const int half = lane >> 4;             // 0: lanes 0-15, 1: lanes 16-31
  const int l15  = lane & 15;
  const int row  = bm * 128 + wv * 16;
  if (row >= M) return;                   // uniform per wave

  // cooperative B loader: thread -> (column-in-tile, 8-element chunk)
  const int nloc  = tid >> 2;             // 0..63
  const int chunk = tid & 3;              // 0..3
  const u16t* Bg  = B + (size_t)(bn * 64 + nloc) * K + chunk * 8;
  const unsigned ldsoff[2] = {
      (unsigned)(size_t)&smem[0][nloc * 32 + chunk * 8],
      (unsigned)(size_t)&smem[1][nloc * 32 + chunk * 8]};

  const u16t* Arow = A + (size_t)(row + l15) * K;
  const int S = K >> 5;                   // K stages of 32

  // ---- prologue: stage 0 in flight ----
  async_ldst_b128(ldsoff[0], Bg);
  FragAB a_cur;
  a_cur.q[0] = *(const uint4*)(Arow + half * 8);        // K = half*8 .. +7
  a_cur.q[1] = *(const uint4*)(Arow + 16 + half * 8);   // K = 16+half*8 .. +7

  v8f acc[4] = {};
  for (int s = 0; s < S; ++s) {
    FragAB a_nxt;
    if (s + 1 < S) {
      async_ldst_b128(ldsoff[(s + 1) & 1], Bg + (s + 1) * 32);
      const u16t* An = Arow + (s + 1) * 32;
      a_nxt.q[0] = *(const uint4*)(An + half * 8);
      a_nxt.q[1] = *(const uint4*)(An + 16 + half * 8);
      if (s + 2 < S) __builtin_prefetch(An + 32, 0, 3);
      asm volatile("s_wait_asynccnt 0x1" ::: "memory");  // stage s landed
    } else {
      asm volatile("s_wait_asynccnt 0x0" ::: "memory");  // drain
    }
    __syncthreads();
    const u16t* bufp = smem[s & 1];
#pragma unroll
    for (int t = 0; t < 4; ++t) {
      FragAB bfr;                          // column frag from LDS (ds_load_b128)
      const u16t* Bp = bufp + (t * 16 + l15) * 32 + half * 8;
      bfr.q[0] = *(const uint4*)(Bp);
      bfr.q[1] = *(const uint4*)(Bp + 16);
      acc[t] = __builtin_amdgcn_wmma_f32_16x16x32_bf16(
          false, a_cur.v, false, bfr.v, (short)0, acc[t], false, false);
    }
    __syncthreads();                       // protect buffer for next overwrite
    if (s + 1 < S) a_cur = a_nxt;
  }

  // ---- epilogue ----
#pragma unroll
  for (int t = 0; t < 4; ++t) {
    const int n = bn * 64 + t * 16 + l15;
    const float bv = bias ? bias[n] : 0.0f;
#pragma unroll
    for (int r = 0; r < 8; ++r) {
      const int m = row + r + half * 8;    // C/D layout: vgpr r <-> M=r / r+8
      float v = acc[t][r] + bv;
      if (act == 1) v = fmaxf(v, 0.0f);
      else if (act == 2) v = softplusf(v);
      C[(size_t)m * N + n] = v;
      if (Cbf) Cbf[(size_t)m * N + n] = f2bf(v);
    }
  }
}

// ---------------- branch token order ----------------
__device__ __forceinline__ int tok_of(int br, int j) {
  if (br == 0) return j;                        // forward
  if (br == 1) return NL - 1 - j;               // backward
  return (j & (NSL - 1)) * LSL + (j >> 4);      // slice perm: (j%16)*256 + j/16
}

// ---------------- causal depthwise conv + SiLU (all 3 branches) ----------------
__global__ __launch_bounds__(256)
void conv_kernel(const float* __restrict__ xz,
                 const float* __restrict__ wf, const float* __restrict__ bf,
                 const float* __restrict__ wb, const float* __restrict__ bb,
                 const float* __restrict__ ws, const float* __restrict__ bs,
                 u16t* __restrict__ xc_bf) {
  long long i = (long long)blockIdx.x * blockDim.x + threadIdx.x;
  if (i >= (long long)3 * NB * NL * NDI) return;
  int d  = (int)(i % NDI);
  int j  = (int)((i / NDI) % NL);
  int b  = (int)((i / ((long long)NDI * NL)) % NB);
  int br = (int)(i / ((long long)NDI * NL * NB));
  const float* w  = (br == 0) ? wf : (br == 1) ? wb : ws;
  const float* bi = (br == 0) ? bf : (br == 1) ? bb : bs;
  float acc = bi[d];
#pragma unroll
  for (int t = 0; t < 4; ++t) {
    int jj = j - 3 + t;
    if (jj >= 0) {
      int tok = tok_of(br, jj);
      acc += w[d * 4 + t] * xz[((size_t)b * NL + tok) * (2 * NDI) + d];
    }
  }
  xc_bf[i] = f2bf(siluf(acc));
}

// ---------------- selective scan: one thread per (branch,b,d) ----------------
__global__ __launch_bounds__(128)
void scan_kernel(const float* __restrict__ dt, const float* __restrict__ dbl,
                 const u16t* __restrict__ xc_bf,
                 const float* __restrict__ Af, const float* __restrict__ Df,
                 const float* __restrict__ Ab, const float* __restrict__ Db,
                 const float* __restrict__ As, const float* __restrict__ Ds,
                 float* __restrict__ y) {
  int tid = blockIdx.x * blockDim.x + threadIdx.x;
  if (tid >= 3 * NB * NDI) return;
  int d  = tid % NDI;
  int b  = (tid / NDI) % NB;
  int br = tid / (NDI * NB);
  const float* Al = ((br == 0) ? Af : (br == 1) ? Ab : As) + (size_t)d * NDS;
  const float  Dv = ((br == 0) ? Df : (br == 1) ? Db : Ds)[d];
  float A[NDS], h[NDS];
#pragma unroll
  for (int s = 0; s < NDS; ++s) { A[s] = -__expf(Al[s]); h[s] = 0.0f; }
  const size_t base = ((size_t)br * NB + b) * NL;
  for (int j = 0; j < NL; ++j) {
    const size_t t = base + j;
    const float dtv = dt[t * NDI + d];
    const float u   = bf2f(xc_bf[t * NDI + d]);
    const float du  = dtv * u;
    const float* bc = dbl + t * 64 + NDTR;        // B at [24,40), C at [40,56)
    float yv = 0.0f;
#pragma unroll
    for (int s = 0; s < NDS; ++s) {
      h[s] = h[s] * __expf(dtv * A[s]) + du * bc[s];
      yv  += h[s] * bc[NDS + s];
    }
    y[t * NDI + d] = yv + u * Dv;
  }
}

// ---------------- gate: gsq=(yf+yb)*silu(z), gsl=ys*silu(z) ----------------
__global__ __launch_bounds__(256)
void gate_kernel(const float* __restrict__ y, const float* __restrict__ xz,
                 u16t* __restrict__ gsq, u16t* __restrict__ gsl) {
  long long i = (long long)blockIdx.x * blockDim.x + threadIdx.x;
  if (i >= (long long)NB * NL * NDI) return;
  int d = (int)(i % NDI);
  int l = (int)((i / NDI) % NL);
  int b = (int)(i / ((long long)NDI * NL));
  float g = siluf(xz[((size_t)b * NL + l) * (2 * NDI) + NDI + d]);
  size_t rf = ((size_t)0 * NB + b) * NL + l;
  size_t rb = ((size_t)1 * NB + b) * NL + (NL - 1 - l);
  int js = (l % LSL) * NSL + l / LSL;             // inverse slice perm
  size_t rs = ((size_t)2 * NB + b) * NL + js;
  float yf = y[rf * NDI + d], yb = y[rb * NDI + d], ysv = y[rs * NDI + d];
  gsq[i] = f2bf((yf + yb) * g);
  gsl[i] = f2bf(ysv * g);
}

// ---------------- out = xsq + xsl (+bf16 copy) ----------------
__global__ __launch_bounds__(256)
void addpack_kernel(const float* __restrict__ a, const float* __restrict__ b,
                    float* __restrict__ o, u16t* __restrict__ obf, int n) {
  int i = blockIdx.x * blockDim.x + threadIdx.x;
  if (i >= n) return;
  float v = a[i] + b[i];
  o[i] = v; obf[i] = f2bf(v);
}

// ---------------- global average pool over tokens ----------------
__global__ __launch_bounds__(256)
void gpool_kernel(const float* __restrict__ out, float* __restrict__ gp) {
  int bc = blockIdx.x;                       // b*NC + c
  int b = bc / NC, c = bc % NC;
  __shared__ float red[256];
  float s = 0.0f;
  for (int l = threadIdx.x; l < NL; l += 256)
    s += out[((size_t)b * NL + l) * NC + c];
  red[threadIdx.x] = s; __syncthreads();
  for (int st = 128; st > 0; st >>= 1) {
    if ((int)threadIdx.x < st) red[threadIdx.x] += red[threadIdx.x + st];
    __syncthreads();
  }
  if (threadIdx.x == 0) gp[bc] = red[0] * (1.0f / NL);
}

// ---------------- tiny pooled MLP: gf = relu(gp@c1^T+c1b)@c2^T + c2b ----------------
__global__ __launch_bounds__(384)
void gf_kernel(const float* __restrict__ gp,
               const float* __restrict__ c1w, const float* __restrict__ c1b,
               const float* __restrict__ c2w, const float* __restrict__ c2b,
               float* __restrict__ gf) {
  __shared__ float hid[NC];
  int c = threadIdx.x, b = blockIdx.x;
  float acc = c1b[c];
  for (int k = 0; k < NC; ++k) acc += c1w[(size_t)c * NC + k] * gp[b * NC + k];
  hid[c] = fmaxf(acc, 0.0f);
  __syncthreads();
  float acc2 = c2b[c];
  for (int k = 0; k < NC; ++k) acc2 += c2w[(size_t)c * NC + k] * hid[k];
  gf[b * NC + c] = acc2;
}

// ---------------- final gated combine + residual, [B,L,C] -> [B,C,L] ----------------
__global__ __launch_bounds__(256)
void final_kernel(const float* __restrict__ xsq, const float* __restrict__ xsl,
                  const float* __restrict__ sp, const float* __restrict__ gf,
                  const float* __restrict__ x, float* __restrict__ out) {
  long long i = (long long)blockIdx.x * blockDim.x + threadIdx.x;
  if (i >= (long long)NB * NL * NC) return;
  int c = (int)(i % NC);
  int l = (int)((i / NC) % NL);
  int b = (int)(i / ((long long)NC * NL));
  size_t tc = ((size_t)b * NL + l) * NC + c;
  float w = sigmoidf(gf[b * NC + c] + sp[tc]);
  size_t oidx = (size_t)b * NC * NL + (size_t)c * NL + l;
  out[oidx] = xsq[tc] * w + (1.0f - w) * xsl[tc] + x[oidx];
}

// ============================= host launcher =============================
static inline void launch_gemm(const u16t* A, const u16t* B, const float* bias,
                               float* C, u16t* Cbf, int M, int N, int K, int act,
                               hipStream_t s) {
  int blocks = ((M + 127) / 128) * (N / 64);
  wmma_gemm_kernel<<<blocks, 256, 0, s>>>(A, B, bias, C, Cbf, M, N, K, act);
}

extern "C" void kernel_launch(void* const* d_in, const int* in_sizes, int n_in,
                              void* d_out, int out_size, void* d_ws, size_t ws_size,
                              hipStream_t stream) {
  (void)in_sizes; (void)n_in; (void)out_size; (void)ws_size;
  const float* x    = (const float*)d_in[0];
  const float* ln_g = (const float*)d_in[1];
  const float* ln_b = (const float*)d_in[2];
  const float* in_w = (const float*)d_in[3];
  const float* outw = (const float*)d_in[4];
  const float* convw[3] = {(const float*)d_in[5],  (const float*)d_in[12], (const float*)d_in[19]};
  const float* convb[3] = {(const float*)d_in[6],  (const float*)d_in[13], (const float*)d_in[20]};
  const float* xpw[3]   = {(const float*)d_in[7],  (const float*)d_in[14], (const float*)d_in[21]};
  const float* dtw[3]   = {(const float*)d_in[8],  (const float*)d_in[15], (const float*)d_in[22]};
  const float* dtb[3]   = {(const float*)d_in[9],  (const float*)d_in[16], (const float*)d_in[23]};
  const float* Alog[3]  = {(const float*)d_in[10], (const float*)d_in[17], (const float*)d_in[24]};
  const float* Dp[3]    = {(const float*)d_in[11], (const float*)d_in[18], (const float*)d_in[25]};
  const float* c1w = (const float*)d_in[26]; const float* c1b = (const float*)d_in[27];
  const float* c2w = (const float*)d_in[28]; const float* c2b = (const float*)d_in[29];
  const float* c3w = (const float*)d_in[30]; const float* c3b = (const float*)d_in[31];
  const float* c4w = (const float*)d_in[32]; const float* c4b = (const float*)d_in[33];
  float* out = (float*)d_out;

  // ---- workspace carve-out ----
  char* base = (char*)d_ws; size_t off = 0;
  auto alloc = [&](size_t bytes) -> void* {
    off = (off + 255) & ~(size_t)255;
    void* p = base + off; off += bytes; return p;
  };
  u16t* w_in_bf  = (u16t*)alloc((size_t)2 * NDI * NC * 2);
  u16t* w_out_bf = (u16t*)alloc((size_t)NC * NDI * 2);
  u16t* w_c3_bf  = (u16t*)alloc((size_t)NC * NC * 2);
  u16t* w_c4_bf  = (u16t*)alloc((size_t)NC * NC * 2);
  u16t* w_xp_bf[3]; for (int t = 0; t < 3; ++t) w_xp_bf[t] = (u16t*)alloc((size_t)64 * NDI * 2);
  u16t* w_dt_bf[3]; for (int t = 0; t < 3; ++t) w_dt_bf[t] = (u16t*)alloc((size_t)NDI * 32 * 2);
  u16t*  xn_bf  = (u16t*)alloc((size_t)NTOK * NC * 2);
  float* xz     = (float*)alloc((size_t)NTOK * 2 * NDI * 4);
  u16t*  xc_bf  = (u16t*)alloc((size_t)3 * NTOK * NDI * 2);
  float* dbl    = (float*)alloc((size_t)3 * NTOK * 64 * 4);
  u16t*  dtr_bf = (u16t*)alloc((size_t)3 * NTOK * 32 * 2);
  float* dt     = (float*)alloc((size_t)3 * NTOK * NDI * 4);
  float* yb     = (float*)alloc((size_t)3 * NTOK * NDI * 4);
  u16t*  gsq_bf = (u16t*)alloc((size_t)NTOK * NDI * 2);
  u16t*  gsl_bf = (u16t*)alloc((size_t)NTOK * NDI * 2);
  float* xsq    = (float*)alloc((size_t)NTOK * NC * 4);
  float* xsl    = (float*)alloc((size_t)NTOK * NC * 4);
  float* osum   = (float*)alloc((size_t)NTOK * NC * 4);
  u16t*  osumbf = (u16t*)alloc((size_t)NTOK * NC * 2);
  float* h1f    = (float*)alloc((size_t)NTOK * NC * 4);
  u16t*  h1bf   = (u16t*)alloc((size_t)NTOK * NC * 2);
  float* sp     = (float*)alloc((size_t)NTOK * NC * 4);
  float* gp     = (float*)alloc((size_t)NB * NC * 4);
  float* gf     = (float*)alloc((size_t)NB * NC * 4);

  // ---- 1) pack weights to bf16 ----
  auto pk = [&](const float* s, u16t* d, int n) {
    pack_bf16_kernel<<<(n + 255) / 256, 256, 0, stream>>>(s, d, n);
  };
  pk(in_w, w_in_bf, 2 * NDI * NC);
  pk(outw, w_out_bf, NC * NDI);
  pk(c3w,  w_c3_bf, NC * NC);
  pk(c4w,  w_c4_bf, NC * NC);
  for (int t = 0; t < 3; ++t) {
    int n1 = 64 * NDI;
    pack_pad_bf16_kernel<<<(n1 + 255) / 256, 256, 0, stream>>>(xpw[t], w_xp_bf[t],
        NDTR + 2 * NDS, NDI, 64, NDI);                 // [56,768] -> [64,768]
    int n2 = NDI * 32;
    pack_pad_bf16_kernel<<<(n2 + 255) / 256, 256, 0, stream>>>(dtw[t], w_dt_bf[t],
        NDI, NDTR, NDI, 32);                           // [768,24] -> [768,32]
  }

  // ---- 2) LayerNorm ----
  layernorm_kernel<<<NTOK, 128, 0, stream>>>(x, ln_g, ln_b, xn_bf);

  // ---- 3) in-projection: xz[8192,1536] = xn @ in_w^T ----
  launch_gemm(xn_bf, w_in_bf, nullptr, xz, nullptr, NTOK, 2 * NDI, NC, 0, stream);

  // ---- 4) conv + SiLU for all 3 branch orderings ----
  {
    long long n = (long long)3 * NTOK * NDI;
    conv_kernel<<<(unsigned)((n + 255) / 256), 256, 0, stream>>>(
        xz, convw[0], convb[0], convw[1], convb[1], convw[2], convb[2], xc_bf);
  }

  // ---- 5) x-projection per branch: dbl[8192,64] = xc @ xpw^T (padded) ----
  for (int t = 0; t < 3; ++t)
    launch_gemm(xc_bf + (size_t)t * NTOK * NDI, w_xp_bf[t], nullptr,
                dbl + (size_t)t * NTOK * 64, nullptr, NTOK, 64, NDI, 0, stream);

  // ---- 6) dt projection per branch: dt = softplus(dt_r @ dtw^T + dtb) ----
  for (int t = 0; t < 3; ++t) {
    int n = NTOK * 32;
    pack_dtr_kernel<<<(n + 255) / 256, 256, 0, stream>>>(
        dbl + (size_t)t * NTOK * 64, dtr_bf + (size_t)t * NTOK * 32, NTOK);
    launch_gemm(dtr_bf + (size_t)t * NTOK * 32, w_dt_bf[t], dtb[t],
                dt + (size_t)t * NTOK * NDI, nullptr, NTOK, NDI, 32, 2, stream);
  }

  // ---- 7) selective scan ----
  {
    int n = 3 * NB * NDI;
    scan_kernel<<<(n + 127) / 128, 128, 0, stream>>>(
        dt, dbl, xc_bf, Alog[0], Dp[0], Alog[1], Dp[1], Alog[2], Dp[2], yb);
  }

  // ---- 8) gate ----
  {
    long long n = (long long)NTOK * NDI;
    gate_kernel<<<(unsigned)((n + 255) / 256), 256, 0, stream>>>(yb, xz, gsq_bf, gsl_bf);
  }

  // ---- 9) out projections ----
  launch_gemm(gsq_bf, w_out_bf, nullptr, xsq, nullptr, NTOK, NC, NDI, 0, stream);
  launch_gemm(gsl_bf, w_out_bf, nullptr, xsl, nullptr, NTOK, NC, NDI, 0, stream);

  // ---- 10) out = xsq + xsl ----
  {
    int n = NTOK * NC;
    addpack_kernel<<<(n + 255) / 256, 256, 0, stream>>>(xsq, xsl, osum, osumbf, n);
  }

  // ---- 11) fusion: pooled MLP branch ----
  gpool_kernel<<<NB * NC, 256, 0, stream>>>(osum, gp);
  gf_kernel<<<NB, NC, 0, stream>>>(gp, c1w, c1b, c2w, c2b, gf);

  // ---- 12) fusion: spatial branch (two 1x1x1 = token GEMMs) ----
  launch_gemm(osumbf, w_c3_bf, c3b, h1f, h1bf, NTOK, NC, NC, 1, stream);
  launch_gemm(h1bf,  w_c4_bf, c4b, sp,  nullptr, NTOK, NC, NC, 0, stream);

  // ---- 13) gated combine + residual, transpose to [B,C,H,W,Z] ----
  {
    long long n = (long long)NTOK * NC;
    final_kernel<<<(unsigned)((n + 255) / 256), 256, 0, stream>>>(
        xsq, xsl, sp, gf, x, out);
  }
}
